// SimpleMamba_82626580840709
// MI455X (gfx1250) — matrix operand order repaired and checked by reading, not compiled, not run on hardware
//
#include <hip/hip_runtime.h>
#include <hip/hip_bf16.h>

// ---------------- problem constants ----------------
#define BATCH   4
#define SEQLEN  2048
#define DMODEL  1024
#define DSTATE  16
#define DCONV   4
#define DINNER  2048
#define MROWS   (BATCH * SEQLEN)          // 8192

typedef __attribute__((ext_vector_type(16))) __bf16 v16bf;
typedef __attribute__((ext_vector_type(8)))  float  v8f;

union FragU { v16bf v; unsigned u[8]; };

__device__ __forceinline__ unsigned short f32_to_bf16(float f) {
    unsigned u = __float_as_uint(f);
    u += 0x7FFFu + ((u >> 16) & 1u);      // round to nearest even
    return (unsigned short)(u >> 16);
}

// ---------------- elementwise f32 -> bf16 ----------------
__global__ void cvt_bf16_kernel(const float* __restrict__ in,
                                unsigned short* __restrict__ out, long n) {
    long i = (long)blockIdx.x * blockDim.x + threadIdx.x;
    if (i < n) out[i] = f32_to_bf16(in[i]);
}

// W is (K x N) row-major f32; out is (N x K) row-major bf16 (W^T)
__global__ void transpose_cvt_kernel(const float* __restrict__ in,
                                     unsigned short* __restrict__ out,
                                     int K, int N) {
    long i = (long)blockIdx.x * blockDim.x + threadIdx.x;
    if (i >= (long)K * N) return;
    int n = (int)(i / K);
    int k = (int)(i - (long)n * K);
    out[i] = f32_to_bf16(in[(long)k * N + n]);
}

// load one K-tile's fragments (A for two M-subtiles + shared B)
__device__ __forceinline__ void load_frags(const unsigned short* arow0,
                                           const unsigned short* arow1,
                                           const unsigned short* brow,
                                           int k0, int half,
                                           FragU& a0, FragU& a1, FragU& b) {
#pragma unroll
    for (int v = 0; v < 8; ++v) {
        // A 16x32 bf16 fragment: lanes0-15 K={0..7,16..23}, lanes16-31 +8
        int ka = k0 + 2 * v + (v >= 4 ? 8 : 0) + 8 * half;
        a0.u[v] = *(const unsigned*)(arow0 + ka);
        a1.u[v] = *(const unsigned*)(arow1 + ka);
        // B 32x16 bf16 fragment: lanes0-15 K={0..7,16..23}, lanes16-31 +8
        int kb = k0 + 2 * v + (v >= 4 ? 8 : 0) + 8 * half;
        b.u[v] = *(const unsigned*)(brow + kb);
    }
}

// ---------------- WMMA bf16 GEMM ----------------
// C[M x N] = A[M x K] (bf16 row-major) * B (given as Bt = B^T, N x K bf16 row-major)
// block = 256 threads (8 waves). Block tile: 256(M) x 16(N).
// Each wave owns TWO 16x16 M-subtiles sharing one B fragment (2 indep. WMMA chains).
// B panel (16 x K bf16) staged into LDS with GLOBAL_LOAD_ASYNC_TO_LDS_B128 (ASYNCcnt).
// K-loop is software-pipelined (ping-pong fragments) so loads of tile k+32 overlap
// the WMMAs of tile k.  K must be a multiple of 64 (holds for all call sites).
// mode 0: out0[m*N+n] = val
// mode 1: split at splitN: n<splitN -> out0[m*splitN+n], else out1[m*splitN+(n-splitN)]
// mode 2: val = softplus(val + bias[n]) -> out0[m*N+n]
__global__ void gemm_bf16_wmma(const unsigned short* __restrict__ A,
                               const unsigned short* __restrict__ Bt,
                               int M, int K, int N, int mode,
                               float* __restrict__ out0,
                               float* __restrict__ out1,
                               const float* __restrict__ bias,
                               int splitN) {
    __shared__ unsigned short sB[16 * 2056];          // 16 x (Kmax=2048 + 8 pad) = 65,792 B

    const int lane = threadIdx.x & 31;
    const int wave = threadIdx.x >> 5;
    const int half = lane >> 4;
    const int col  = lane & 15;
    const int n0 = blockIdx.x * 16;
    const int m0 = blockIdx.y * 256 + wave * 32;      // two 16-row subtiles per wave
    const int KP = K + 8;                              // padded LDS row stride (shorts), 16B-aligned rows

    // ---- async copy of the 16 x K bf16 B panel into LDS (B128 chunks, ASYNCcnt) ----
    {
        const unsigned ldsBase = (unsigned)(unsigned long long)(&sB[0]);
        const int cpr = K >> 3;                        // 16-byte chunks per row
        for (int j = threadIdx.x; j < 16 * cpr; j += 256) {
            int row = j / cpr;
            int c   = j - row * cpr;
            unsigned ldsOff = ldsBase + (unsigned)(row * KP * 2 + c * 16);
            unsigned long long ga =
                (unsigned long long)(Bt + (long)(n0 + row) * K + (c << 3));
            asm volatile("global_load_async_to_lds_b128 %0, %1, off"
                         :: "v"(ldsOff), "v"(ga) : "memory");
        }
        asm volatile("s_wait_asynccnt 0" ::: "memory");
    }
    __syncthreads();

    const unsigned short* arow0 = A + (long)(m0 + col) * K;
    const unsigned short* arow1 = A + (long)(m0 + 16 + col) * K;
    const unsigned short* brow  = &sB[col * KP];

    v8f c0 = {};
    v8f c1 = {};

    // 2-stage software pipeline over K (K % 64 == 0)
    FragU pa0, pa1, pb;                                // ping
    FragU qa0, qa1, qb;                                // pong
    load_frags(arow0, arow1, brow, 0, half, pa0, pa1, pb);
    for (int k0 = 0; k0 + 32 < K; k0 += 64) {
        // issue loads for tile k0+32 before consuming tile k0
        load_frags(arow0, arow1, brow, k0 + 32, half, qa0, qa1, qb);
        __builtin_prefetch(arow0 + k0 + 64, 0, 1);     // -> global_prefetch_b8
        __builtin_prefetch(arow1 + k0 + 64, 0, 1);
        c0 = __builtin_amdgcn_wmma_f32_16x16x32_bf16(
                 false, pa0.v, false, pb.v, (short)0, c0, false, false);
        c1 = __builtin_amdgcn_wmma_f32_16x16x32_bf16(
                 false, pa1.v, false, pb.v, (short)0, c1, false, false);
        if (k0 + 64 < K)                                // uniform guard, no tail load
            load_frags(arow0, arow1, brow, k0 + 64, half, pa0, pa1, pb);
        c0 = __builtin_amdgcn_wmma_f32_16x16x32_bf16(
                 false, qa0.v, false, qb.v, (short)0, c0, false, false);
        c1 = __builtin_amdgcn_wmma_f32_16x16x32_bf16(
                 false, qa1.v, false, qb.v, (short)0, c1, false, false);
    }

    // epilogue: C layout -> m = v + 8*half, n = col
#pragma unroll
    for (int t = 0; t < 2; ++t) {
        v8f c = t ? c1 : c0;
        int mb = m0 + t * 16;
#pragma unroll
        for (int v = 0; v < 8; ++v) {
            int m = mb + v + 8 * half;
            int n = n0 + col;
            float val = c[v];
            if (mode == 2) {
                val += bias[n];
                val = (val > 20.f) ? val : log1pf(expf(val));  // softplus
            }
            if (mode == 1) {
                if (n < splitN) out0[(long)m * splitN + n] = val;
                else            out1[(long)m * splitN + (n - splitN)] = val;
            } else {
                out0[(long)m * N + n] = val;
            }
        }
    }
}

// ---------------- causal depthwise conv1d + SiLU ----------------
__global__ void conv_silu_kernel(const float* __restrict__ xc,
                                 const float* __restrict__ cw,
                                 const float* __restrict__ cb,
                                 float* __restrict__ xs,
                                 unsigned short* __restrict__ xsb) {
    long idx = (long)blockIdx.x * blockDim.x + threadIdx.x;   // over MROWS*DINNER
    int  d  = (int)(idx & (DINNER - 1));
    long ml = idx >> 11;                                      // b*L + l
    int  l  = (int)(ml & (SEQLEN - 1));
    float acc = cb[d];
#pragma unroll
    for (int t = 0; t < DCONV; ++t) {
        int ll = l - (DCONV - 1) + t;
        if (ll >= 0)
            acc += cw[d * DCONV + t] * xc[(ml - (long)(DCONV - 1 - t)) * DINNER + d];
    }
    float sig = 1.f / (1.f + expf(-acc));
    float v = acc * sig;
    xs[idx]  = v;
    xsb[idx] = f32_to_bf16(v);
}

// ---------------- selective scan (sequential over L) ----------------
// grid = BATCH * (DINNER/256) blocks, 256 threads; one thread = one channel d.
__global__ void scan_kernel(const float* __restrict__ delta,
                            const float* __restrict__ BC,     // [B,L,2*DSTATE]
                            const float* __restrict__ xs,
                            const float* __restrict__ zg,
                            const float* __restrict__ A_log,
                            const float* __restrict__ Dp,
                            unsigned short* __restrict__ yb) {
    const int chunks = DINNER / 256;                          // 8
    int b = blockIdx.x / chunks;
    int d = (blockIdx.x % chunks) * 256 + threadIdx.x;

    float Ar[DSTATE], h[DSTATE];
#pragma unroll
    for (int s = 0; s < DSTATE; ++s) {
        Ar[s] = -expf(A_log[d * DSTATE + s]);
        h[s]  = 0.f;
    }
    float Dv = Dp[d];

    __shared__ float sBC[2 * DSTATE];
    long base = (long)b * SEQLEN;
    for (int l = 0; l < SEQLEN; ++l) {
        if (threadIdx.x < 2 * DSTATE)
            sBC[threadIdx.x] = BC[(base + l) * (2 * DSTATE) + threadIdx.x];
        __syncthreads();

        long idx = (base + l) * DINNER + d;
        float dt  = delta[idx];
        float xv  = xs[idx];
        float zv  = zg[idx];
        float xdt = xv * dt;
        float y   = Dv * xv;
#pragma unroll
        for (int s = 0; s < DSTATE; ++s) {
            float dA = 1.f + Ar[s] * dt;                      // Euler discretization
            h[s] = h[s] * dA + sBC[s] * xdt;
            y   += h[s] * sBC[DSTATE + s];
        }
        float sig = 1.f / (1.f + expf(-zv));
        y *= zv * sig;                                        // * silu(z)
        yb[idx] = f32_to_bf16(y);
        __syncthreads();
    }
}

// ---------------- host launcher ----------------
extern "C" void kernel_launch(void* const* d_in, const int* in_sizes, int n_in,
                              void* d_out, int out_size, void* d_ws, size_t ws_size,
                              hipStream_t stream) {
    (void)in_sizes; (void)n_in; (void)out_size; (void)ws_size;
    const float* x       = (const float*)d_in[0];
    const float* W_in    = (const float*)d_in[1];
    const float* conv_w  = (const float*)d_in[2];
    const float* conv_b  = (const float*)d_in[3];
    const float* W_xproj = (const float*)d_in[4];
    const float* W_dt    = (const float*)d_in[5];
    const float* b_dt    = (const float*)d_in[6];
    const float* A_log   = (const float*)d_in[7];
    const float* D_param = (const float*)d_in[8];
    const float* W_out   = (const float*)d_in[9];
    float* out = (float*)d_out;

    char* ws = (char*)d_ws;
    size_t off = 0;
    auto alloc = [&](size_t bytes) -> void* {
        void* p = ws + off;
        off += (bytes + 255) & ~(size_t)255;
        return p;
    };

    unsigned short* xb    = (unsigned short*)alloc((size_t)MROWS * DMODEL * 2);
    unsigned short* WinT  = (unsigned short*)alloc((size_t)(2 * DINNER) * DMODEL * 2);
    unsigned short* WdtT  = (unsigned short*)alloc((size_t)DINNER * DINNER * 2);
    unsigned short* WxT   = (unsigned short*)alloc((size_t)(2 * DSTATE) * DINNER * 2);
    unsigned short* WoT   = (unsigned short*)alloc((size_t)DMODEL * DINNER * 2);
    float* xc    = (float*)alloc((size_t)MROWS * DINNER * 4);
    float* zbuf  = (float*)alloc((size_t)MROWS * DINNER * 4);
    float* xs    = (float*)alloc((size_t)MROWS * DINNER * 4);
    unsigned short* xsb = (unsigned short*)alloc((size_t)MROWS * DINNER * 2);
    float* delta = (float*)alloc((size_t)MROWS * DINNER * 4);
    float* BC    = (float*)alloc((size_t)MROWS * 2 * DSTATE * 4);
    unsigned short* yb  = (unsigned short*)alloc((size_t)MROWS * DINNER * 2);

    // 1) convert activations / weights to bf16 (weights transposed to N x K)
    {
        long n = (long)MROWS * DMODEL;
        cvt_bf16_kernel<<<(unsigned)((n + 255) / 256), 256, 0, stream>>>(x, xb, n);
    }
    transpose_cvt_kernel<<<(unsigned)(((long)DMODEL * 2 * DINNER + 255) / 256), 256, 0, stream>>>(
        W_in, WinT, DMODEL, 2 * DINNER);
    transpose_cvt_kernel<<<(unsigned)(((long)DINNER * DINNER + 255) / 256), 256, 0, stream>>>(
        W_dt, WdtT, DINNER, DINNER);
    transpose_cvt_kernel<<<(unsigned)(((long)DINNER * 2 * DSTATE + 255) / 256), 256, 0, stream>>>(
        W_xproj, WxT, DINNER, 2 * DSTATE);
    transpose_cvt_kernel<<<(unsigned)(((long)DINNER * DMODEL + 255) / 256), 256, 0, stream>>>(
        W_out, WoT, DINNER, DMODEL);

    // 2) xz = x @ W_in  -> split into xc | z
    gemm_bf16_wmma<<<dim3((2 * DINNER) / 16, MROWS / 256), 256, 0, stream>>>(
        xb, WinT, MROWS, DMODEL, 2 * DINNER, /*mode=*/1, xc, zbuf, nullptr, DINNER);

    // 3) causal depthwise conv + SiLU -> xs (f32 + bf16)
    conv_silu_kernel<<<(unsigned)(((long)MROWS * DINNER) / 256), 256, 0, stream>>>(
        xc, conv_w, conv_b, xs, xsb);

    // 4) delta = softplus(xs @ W_dt + b_dt)
    gemm_bf16_wmma<<<dim3(DINNER / 16, MROWS / 256), 256, 0, stream>>>(
        xsb, WdtT, MROWS, DINNER, DINNER, /*mode=*/2, delta, nullptr, b_dt, 0);

    // 5) [B|C] = xs @ W_xproj
    gemm_bf16_wmma<<<dim3((2 * DSTATE) / 16, MROWS / 256), 256, 0, stream>>>(
        xsb, WxT, MROWS, DINNER, 2 * DSTATE, /*mode=*/0, BC, nullptr, nullptr, 0);

    // 6) selective scan + SiLU(z) gating -> y (bf16)
    scan_kernel<<<BATCH * (DINNER / 256), 256, 0, stream>>>(
        delta, BC, xs, zbuf, A_log, D_param, yb);

    // 7) out = y @ W_out
    gemm_bf16_wmma<<<dim3(DMODEL / 16, MROWS / 256), 256, 0, stream>>>(
        yb, WoT, MROWS, DINNER, DMODEL, /*mode=*/0, out, nullptr, nullptr, 0);
}
